// GAT_86388972191751
// MI455X (gfx1250) — compile-verified, matrix operand order checked by reading, not run on hardware
//
#include <hip/hip_runtime.h>
#include <hip/hip_bf16.h>

// GAT forward for gfx1250 (MI455X).
// GEMMs (encoder, per-layer h = X@W_att, decoder) run on v_wmma_f32_16x16x32_bf16
// (bf16 inputs, f32 accumulate). Edge softmax/aggregation kept in f32; the whole
// per-layer working set (~150MB) is L2-resident on the 192MB L2, so the scatter
// phase is L2-atomic bound, not HBM bound.

#define NFEAT   256
#define NHID    64
#define NHEADS  4
#define NHC     (NHEADS * NHID)   // 256
#define NCLASS  40
#define NCLASSP 48                // padded to 3 WMMA tiles
#define NLAYERS 4
#define NEG_SLOPE 0.2f

typedef __attribute__((ext_vector_type(16))) __bf16 v16bf;
typedef __attribute__((ext_vector_type(8)))  float  v8f;

static __device__ __forceinline__ unsigned short f2bf(float f) {
  unsigned int u = __float_as_uint(f);
  u += 0x7FFFu + ((u >> 16) & 1u);       // round-to-nearest-even
  return (unsigned short)(u >> 16);
}

// ---------------- conversion / prep kernels ----------------

__global__ void k_f32_to_bf16(const float* __restrict__ in,
                              unsigned short* __restrict__ out, int n) {
  int i = blockIdx.x * blockDim.x + threadIdx.x;
  if (i < n) out[i] = f2bf(in[i]);
}

// in [R,C] f32 row-major -> out [C,R] bf16 (i.e. B stored column-major)
__global__ void k_transpose_bf16(const float* __restrict__ in,
                                 unsigned short* __restrict__ out, int R, int C) {
  int i = blockIdx.x * blockDim.x + threadIdx.x;
  if (i >= R * C) return;
  int r = i / C, c = i % C;
  out[(size_t)c * R + r] = f2bf(in[i]);
}

// W_dec [64,40] -> WdecT [48,64] bf16, rows 40..47 zero
__global__ void k_pad_transpose_wdec(const float* __restrict__ in,
                                     unsigned short* __restrict__ out) {
  int i = blockIdx.x * blockDim.x + threadIdx.x;
  if (i >= NCLASSP * NHID) return;
  int n = i / NHID, k = i % NHID;
  out[i] = f2bf((n < NCLASS) ? in[k * NCLASS + n] : 0.f);
}

__global__ void k_pad_bias(const float* __restrict__ in, float* __restrict__ out,
                           int n, int npad) {
  int i = blockIdx.x * blockDim.x + threadIdx.x;
  if (i < npad) out[i] = (i < n) ? in[i] : 0.f;
}

// ---------------- WMMA GEMM: C[M,Nstore] = act(A[M,K] @ B + bias) ----------------
// A: bf16 row-major [M,K]; Bt: bf16 [Nm,K] (B column-major). One wave per 16x16 tile.
__global__ void k_wmma_gemm(const unsigned short* __restrict__ A,
                            const unsigned short* __restrict__ Bt,
                            const float* __restrict__ bias,
                            float* __restrict__ C,
                            int M, int Nm, int K, int Nstore, int relu) {
  const int wave = (int)((blockIdx.x * (unsigned)blockDim.x + threadIdx.x) >> 5);
  const int lane = threadIdx.x & 31;
  const int tilesN = Nm >> 4;
  const int tiles  = (M >> 4) * tilesN;
  if (wave >= tiles) return;                 // wave-uniform: EXEC all-1s below
  const int tm = wave / tilesN, tn = wave % tilesN;
  const int g   = lane >> 4;                 // half-wave select
  const int r16 = lane & 15;

  // 16-bit A-matrix fragment K-offsets (ISA 7.12.2), +8 for lanes 16..31
  const int kof[8] = {0, 2, 4, 6, 16, 18, 20, 22};
  const unsigned short* Arow = A  + (size_t)(tm * 16 + r16) * K + g * 8;
  const unsigned short* Brow = Bt + (size_t)(tn * 16 + r16) * K + g * 8;

  v8f acc = {};
  for (int kk = 0; kk < K; kk += 32) {
    union { v16bf v; unsigned int u[8]; } a, b;
#pragma unroll
    for (int r = 0; r < 8; ++r) {
      a.u[r] = *(const unsigned int*)(Arow + kk + kof[r]);
      b.u[r] = *(const unsigned int*)(Brow + kk + kof[r]);
    }
    acc = __builtin_amdgcn_wmma_f32_16x16x32_bf16(false, a.v, false, b.v,
                                                  (short)0, acc, false, false);
  }

  const int col = tn * 16 + r16;
  if (col >= Nstore) return;                 // only affects stores (WMMA done)
  const float bv = bias ? bias[col] : 0.f;
  const int row0 = tm * 16 + g * 8;          // C/D layout: vgpr r -> row0 + r
#pragma unroll
  for (int r = 0; r < 8; ++r) {
    float v = acc[r] + bv;
    if (relu) v = fmaxf(v, 0.f);
    C[(size_t)(row0 + r) * Nstore + col] = v;
  }
}

// ---------------- attention scores: a_src/a_dst [N,4] ----------------

__global__ void k_scores(const float* __restrict__ h,
                         const float* __restrict__ att_src,
                         const float* __restrict__ att_dst,
                         float* __restrict__ asrc, float* __restrict__ adst, int N) {
  int t = blockIdx.x * blockDim.x + threadIdx.x;
  if (t >= N * NHEADS) return;
  int n = t >> 2, hd = t & 3;
  const float* hp = h + (size_t)n * NHC + hd * NHID;
  const float* ws = att_src + hd * NHID;
  const float* wd = att_dst + hd * NHID;
  float s1 = 0.f, s2 = 0.f;
#pragma unroll 8
  for (int c = 0; c < NHID; ++c) { float v = hp[c]; s1 += v * ws[c]; s2 += v * wd[c]; }
  asrc[t] = s1; adst[t] = s2;
}

// ---------------- per-layer init (m=-inf, s=0, acc=0) ----------------

__global__ void k_init(float* __restrict__ m, float* __restrict__ s,
                       float* __restrict__ acc, int N) {
  int i = blockIdx.x * blockDim.x + threadIdx.x;
  if (i < N * NHC) acc[i] = 0.f;
  if (i < N * NHEADS) { m[i] = -__builtin_inff(); s[i] = 0.f; }
}

// float atomic max via signed/unsigned int atomics (valid with -inf init, no NaNs)
static __device__ __forceinline__ void atomicMaxF(float* addr, float val) {
  if (val >= 0.f) atomicMax((int*)addr, __float_as_int(val));
  else            atomicMin((unsigned int*)addr, (unsigned int)__float_as_int(val));
}

// ---------------- edge pass 1: segment max ----------------

__global__ void k_edge_max(const int* __restrict__ ei,
                           const float* __restrict__ asrc,
                           const float* __restrict__ adst,
                           float* __restrict__ m, int E, int N) {
  int t = blockIdx.x * blockDim.x + threadIdx.x;
  int ET = E + N;
  if (t >= ET * NHEADS) return;
  int e = t >> 2, hd = t & 3;
  int src = (e < E) ? ei[e]     : (e - E);   // self loops appended
  int dst = (e < E) ? ei[E + e] : (e - E);
  float v = asrc[src * NHEADS + hd] + adst[dst * NHEADS + hd];
  v = (v > 0.f) ? v : NEG_SLOPE * v;         // leaky relu
  atomicMaxF(&m[dst * NHEADS + hd], v);
}

// ---------------- edge pass 2: exp + segment sum ----------------

__global__ void k_edge_exp(const int* __restrict__ ei,
                           const float* __restrict__ asrc,
                           const float* __restrict__ adst,
                           const float* __restrict__ m,
                           float* __restrict__ s, float* __restrict__ eexp,
                           int E, int N) {
  int t = blockIdx.x * blockDim.x + threadIdx.x;
  int ET = E + N;
  if (t >= ET * NHEADS) return;
  int e = t >> 2, hd = t & 3;
  int src = (e < E) ? ei[e]     : (e - E);
  int dst = (e < E) ? ei[E + e] : (e - E);
  float v = asrc[src * NHEADS + hd] + adst[dst * NHEADS + hd];
  v = (v > 0.f) ? v : NEG_SLOPE * v;
  float ex = __expf(v - m[dst * NHEADS + hd]);
  eexp[t] = ex;
  unsafeAtomicAdd(&s[dst * NHEADS + hd], ex);   // global_atomic_add_f32
}

// ---------------- edge pass 3: acc[dst] += alpha * h[src], one wave per edge ----
// Lane owns 8 contiguous channels (all within one head -> single alpha per lane).

__global__ void k_edge_aggr(const int* __restrict__ ei,
                            const float* __restrict__ h,
                            const float* __restrict__ eexp,
                            const float* __restrict__ s,
                            float* __restrict__ acc, int E, int N) {
  int wid  = (int)((blockIdx.x * (unsigned)blockDim.x + threadIdx.x) >> 5);
  int lane = threadIdx.x & 31;
  int ET = E + N;
  if (wid >= ET) return;
  int src = (wid < E) ? ei[wid]     : (wid - E);
  int dst = (wid < E) ? ei[E + wid] : (wid - E);
  int hd = lane >> 3;                         // channels lane*8..lane*8+7 -> head lane/8
  float alpha = eexp[(size_t)wid * NHEADS + hd] / s[dst * NHEADS + hd];
  const float* hp = h   + (size_t)src * NHC + lane * 8;
  float*       ap = acc + (size_t)dst * NHC + lane * 8;
  float4 v0 = *(const float4*)(hp);
  float4 v1 = *(const float4*)(hp + 4);
  unsafeAtomicAdd(ap + 0, alpha * v0.x);
  unsafeAtomicAdd(ap + 1, alpha * v0.y);
  unsafeAtomicAdd(ap + 2, alpha * v0.z);
  unsafeAtomicAdd(ap + 3, alpha * v0.w);
  unsafeAtomicAdd(ap + 4, alpha * v1.x);
  unsafeAtomicAdd(ap + 5, alpha * v1.y);
  unsafeAtomicAdd(ap + 6, alpha * v1.z);
  unsafeAtomicAdd(ap + 7, alpha * v1.w);
}

// ---------------- combine: X += mean over groups of 4 of relu(acc + b_att) ------
// conv.reshape(n, NHID, NHEADS).mean(-1): flat groups {4c',..,4c'+3}.

__global__ void k_combine(const float* __restrict__ acc,
                          const float* __restrict__ b_att,
                          float* __restrict__ X, int N) {
  int t = blockIdx.x * blockDim.x + threadIdx.x;
  if (t >= N * NHID) return;
  int n = t >> 6, cp = t & 63;
  float4 a4 = *(const float4*)(acc + (size_t)n * NHC + cp * 4);
  float4 b4 = *(const float4*)(b_att + cp * 4);
  float v = fmaxf(a4.x + b4.x, 0.f) + fmaxf(a4.y + b4.y, 0.f) +
            fmaxf(a4.z + b4.z, 0.f) + fmaxf(a4.w + b4.w, 0.f);
  X[t] += 0.25f * v;
}

// ---------------- host ----------------

static inline int cdiv(long long a, long long b) { return (int)((a + b - 1) / b); }

extern "C" void kernel_launch(void* const* d_in, const int* in_sizes, int n_in,
                              void* d_out, int out_size, void* d_ws, size_t ws_size,
                              hipStream_t stream) {
  const float* x       = (const float*)d_in[0];
  const int*   ei      = (const int*)  d_in[1];
  const float* W_enc   = (const float*)d_in[2];
  const float* b_enc   = (const float*)d_in[3];
  const float* W_att   = (const float*)d_in[4];
  const float* att_src = (const float*)d_in[5];
  const float* att_dst = (const float*)d_in[6];
  const float* b_att   = (const float*)d_in[7];
  const float* W_dec   = (const float*)d_in[8];
  const float* b_dec   = (const float*)d_in[9];

  const int N = in_sizes[0] / NFEAT;   // 50000 (multiple of 16)
  const int E = in_sizes[1] / 2;       // 800000
  const int ET = E + N;

  // workspace carve (~165 MB)
  char* p = (char*)d_ws;
  auto alloc = [&](size_t bytes) -> char* {
    char* r = p; p += (bytes + 255) & ~(size_t)255; return r;
  };
  unsigned short* x_bf  = (unsigned short*)alloc((size_t)N * NFEAT * 2);
  unsigned short* WencT = (unsigned short*)alloc((size_t)NHID * NFEAT * 2);   // [64,256]
  unsigned short* WattT = (unsigned short*)alloc((size_t)NHC * NHID * 2);     // [256,64]
  unsigned short* WdecT = (unsigned short*)alloc((size_t)NCLASSP * NHID * 2); // [48,64]
  float*          bdecp = (float*)alloc(NCLASSP * 4);
  float*          Xcur  = (float*)alloc((size_t)N * NHID * 4);
  unsigned short* Xbf   = (unsigned short*)alloc((size_t)N * NHID * 2);
  float*          hfeat = (float*)alloc((size_t)N * NHC * 4);
  float*          asrc  = (float*)alloc((size_t)N * NHEADS * 4);
  float*          adst  = (float*)alloc((size_t)N * NHEADS * 4);
  float*          mbuf  = (float*)alloc((size_t)N * NHEADS * 4);
  float*          sbuf  = (float*)alloc((size_t)N * NHEADS * 4);
  float*          eexp  = (float*)alloc((size_t)ET * NHEADS * 4);
  float*          accb  = (float*)alloc((size_t)N * NHC * 4);

  const int B = 256;   // 8 waves per block

  // weight prep (redone every call: deterministic, inputs never mutated)
  k_f32_to_bf16<<<cdiv((size_t)N * NFEAT, B), B, 0, stream>>>(x, x_bf, N * NFEAT);
  k_transpose_bf16<<<cdiv(NFEAT * NHID, B), B, 0, stream>>>(W_enc, WencT, NFEAT, NHID);
  k_transpose_bf16<<<cdiv(NHID * NHC, B), B, 0, stream>>>(W_att, WattT, NHID, NHC);
  k_pad_transpose_wdec<<<cdiv(NCLASSP * NHID, B), B, 0, stream>>>(W_dec, WdecT);
  k_pad_bias<<<1, B, 0, stream>>>(b_dec, bdecp, NCLASS, NCLASSP);

  // encoder: Xcur = relu(x @ W_enc + b_enc)
  {
    int tiles = (N / 16) * (NHID / 16);
    k_wmma_gemm<<<cdiv(tiles, 8), B, 0, stream>>>(x_bf, WencT, b_enc, Xcur,
                                                  N, NHID, NFEAT, NHID, 1);
  }

  for (int layer = 0; layer < NLAYERS; ++layer) {
    k_f32_to_bf16<<<cdiv((size_t)N * NHID, B), B, 0, stream>>>(Xcur, Xbf, N * NHID);
    // h = X @ W_att   [N,256]
    int tiles = (N / 16) * (NHC / 16);
    k_wmma_gemm<<<cdiv(tiles, 8), B, 0, stream>>>(Xbf, WattT, nullptr, hfeat,
                                                  N, NHC, NHID, NHC, 0);
    k_scores<<<cdiv((size_t)N * NHEADS, B), B, 0, stream>>>(hfeat, att_src, att_dst,
                                                            asrc, adst, N);
    k_init<<<cdiv((size_t)N * NHC, B), B, 0, stream>>>(mbuf, sbuf, accb, N);
    k_edge_max<<<cdiv((size_t)ET * NHEADS, B), B, 0, stream>>>(ei, asrc, adst, mbuf, E, N);
    k_edge_exp<<<cdiv((size_t)ET * NHEADS, B), B, 0, stream>>>(ei, asrc, adst, mbuf,
                                                               sbuf, eexp, E, N);
    k_edge_aggr<<<cdiv(ET, 8), B, 0, stream>>>(ei, hfeat, eexp, sbuf, accb, E, N);
    k_combine<<<cdiv((size_t)N * NHID, B), B, 0, stream>>>(accb, b_att, Xcur, N);
  }

  // decoder: out = X @ W_dec + b_dec   (N padded 40->48, store-guarded)
  k_f32_to_bf16<<<cdiv((size_t)N * NHID, B), B, 0, stream>>>(Xcur, Xbf, N * NHID);
  {
    int tiles = (N / 16) * (NCLASSP / 16);
    k_wmma_gemm<<<cdiv(tiles, 8), B, 0, stream>>>(Xbf, WdecT, bdecp, (float*)d_out,
                                                  N, NCLASSP, NHID, NCLASS, 0);
  }
}
